// SAGE_3066606649551
// MI455X (gfx1250) — matrix operand order repaired and checked by reading, not compiled, not run on hardware
//
#include <hip/hip_runtime.h>
#include <stdint.h>

typedef __attribute__((ext_vector_type(16))) __bf16 v16bf;
typedef __attribute__((ext_vector_type(8)))  float  v8f;
typedef __attribute__((ext_vector_type(4)))  float  f32x4;
typedef __attribute__((ext_vector_type(4)))  unsigned int u32x4;
typedef __attribute__((ext_vector_type(2)))  unsigned int u32x2;

union FragU { u32x4 u[2]; v16bf v; };

__device__ __forceinline__ unsigned short f32_to_bf16(float f) {
    union { float f; unsigned u; } c; c.f = f;
    unsigned r = c.u + 0x7FFFu + ((c.u >> 16) & 1u);   // round-to-nearest-even
    return (unsigned short)(r >> 16);
}
__device__ __forceinline__ float bf16_to_f32(unsigned short h) {
    union { unsigned u; float f; } c; c.u = ((unsigned)h) << 16;
    return c.f;
}
__device__ __forceinline__ unsigned pack2_bf16(float a, float b) {
    return (unsigned)f32_to_bf16(a) | ((unsigned)f32_to_bf16(b) << 16);
}
__device__ __forceinline__ float lo_bf16(unsigned w) {
    union { unsigned u; float f; } c; c.u = w << 16; return c.f;
}
__device__ __forceinline__ float hi_bf16(unsigned w) {
    union { unsigned u; float f; } c; c.u = w & 0xFFFF0000u; return c.f;
}

#define NT  16      // padded N tiles: 16*16 = 256 output columns
#define TPB 256     // 8 waves of 32

// Pack [Wself ; Wneigh] (fp32, row-major (K=2*din, dout)) into bf16 B-fragment
// layout: halves[(((kt*NT + nt)*32 + lane)*16) + e] = W[kt*32 + (lane>>4)*16 + e][nt*16 + (lane&15)]
// zero-padded for n >= dout.
__global__ void pack_weights_kernel(const float* __restrict__ Wself,
                                    const float* __restrict__ Wneigh,
                                    unsigned short* __restrict__ wfrag,
                                    int din, int dout)
{
    const int K  = 2 * din;
    const int KT = K >> 5;
    const size_t total = (size_t)KT * NT * 32 * 16;
    for (size_t idx = (size_t)blockIdx.x * blockDim.x + threadIdx.x; idx < total;
         idx += (size_t)gridDim.x * blockDim.x) {
        int e    = (int)(idx & 15);
        int lane = (int)((idx >> 4) & 31);
        size_t rem = idx >> 9;          // / (32*16)
        int nt = (int)(rem & (NT - 1)); // NT == 16
        int kt = (int)(rem >> 4);
        int k = kt * 32 + (lane >> 4) * 16 + e;
        int n = nt * 16 + (lane & 15);
        float v = 0.0f;
        if (n < dout)
            v = (k < din) ? Wself[(size_t)k * dout + n]
                          : Wneigh[(size_t)(k - din) * dout + n];
        wfrag[idx] = f32_to_bf16(v);
    }
}

// One block = 16 destination nodes. Vectorized (b128/b64) gather of self + mean
// rows into an LDS bf16 A-tile, then 8 waves each compute two 16x16 output
// tiles of [self|mean] @ [Wself;Wneigh] via v_wmma_f32_16x16x32_bf16.
template <int DIN, int DOUT, int FANOUT, bool IN_BF16, bool OUT_F32, bool RELU>
__global__ __launch_bounds__(TPB)
void sage_layer_kernel(const void* __restrict__ hin_v,
                       const int*  __restrict__ neigh,
                       const unsigned short* __restrict__ wfrag,
                       const float* __restrict__ bias,
                       void* __restrict__ hout_v)
{
    constexpr int K     = 2 * DIN;
    constexpr int KT    = K >> 5;        // 16x16x32 steps
    constexpr int PITCH = 2 * DIN + 8;   // halves; rows 16B-aligned, dword stride % 64 == 4
    constexpr int DV    = DIN / 4;       // vec4 chunks per row (power of 2)
    constexpr float INV_FANOUT = 1.0f / (float)FANOUT;

    __shared__ alignas(16) unsigned short As[16][PITCH];
    __shared__ int nidx[16 * FANOUT];

    const int tid  = threadIdx.x;
    const int base = (int)blockIdx.x * 16;

    const float*          hin_f = (const float*)hin_v;
    const unsigned short* hin_b = (const unsigned short*)hin_v;

    // stage neighbor indices for this block's 16 nodes
    if (tid < 16 * FANOUT) {
        int r = tid / FANOUT, f = tid - r * FANOUT;
        nidx[tid] = neigh[(size_t)(base + r) * FANOUT + f];
    }
    __syncthreads();

    // self rows -> As[r][0..DIN), 4 elements per thread-iteration
    for (int idx = tid; idx < 16 * DV; idx += TPB) {
        int r = idx / DV, kv = idx & (DV - 1);
        size_t off = (size_t)(base + r) * DIN + kv * 4;
        u32x2 packed;
        if (IN_BF16) {
            packed = *(const u32x2*)(hin_b + off);        // raw bf16 copy
        } else {
            f32x4 v = *(const f32x4*)(hin_f + off);
            packed.x = pack2_bf16(v.x, v.y);
            packed.y = pack2_bf16(v.z, v.w);
        }
        *(u32x2*)&As[r][kv * 4] = packed;
    }
    // mean rows -> As[r][DIN..2*DIN)
    for (int idx = tid; idx < 16 * DV; idx += TPB) {
        int r = idx / DV, kv = idx & (DV - 1);
        float s0 = 0.f, s1 = 0.f, s2 = 0.f, s3 = 0.f;
        for (int f = 0; f < FANOUT; ++f) {
            size_t off = (size_t)nidx[r * FANOUT + f] * DIN + kv * 4;
            if (IN_BF16) {
                u32x2 w = *(const u32x2*)(hin_b + off);
                s0 += lo_bf16(w.x); s1 += hi_bf16(w.x);
                s2 += lo_bf16(w.y); s3 += hi_bf16(w.y);
            } else {
                f32x4 v = *(const f32x4*)(hin_f + off);
                s0 += v.x; s1 += v.y; s2 += v.z; s3 += v.w;
            }
        }
        u32x2 packed;
        packed.x = pack2_bf16(s0 * INV_FANOUT, s1 * INV_FANOUT);
        packed.y = pack2_bf16(s2 * INV_FANOUT, s3 * INV_FANOUT);
        *(u32x2*)&As[r][DIN + kv * 4] = packed;
    }
    __syncthreads();

    const int lane = tid & 31;
    const int wave = tid >> 5;
    const int sel  = lane >> 4;   // K-half / M-half selector
    const int l15  = lane & 15;
    const int t0 = wave, t1 = wave + 8;   // this wave's two N tiles (NT=16, always valid)

    v8f acc0 = {};
    v8f acc1 = {};

    for (int kt = 0; kt < KT; ++kt) {
        // A fragment in contiguous K order (matches packed B):
        // lane row m=l15; elems 0..15 = K kt*32+sel*16..
        FragU a;
        a.u[0] = *(const u32x4*)&As[l15][kt * 32 + sel * 16];
        a.u[1] = *(const u32x4*)&As[l15][kt * 32 + sel * 16 + 8];

        // B fragments: coalesced 2x128-bit loads from pre-packed weights (L2-resident)
        FragU b0, b1;
        const u32x4* bp0 = (const u32x4*)&wfrag[(((size_t)kt * NT + t0) * 32 + lane) * 16];
        const u32x4* bp1 = (const u32x4*)&wfrag[(((size_t)kt * NT + t1) * 32 + lane) * 16];
        b0.u[0] = bp0[0]; b0.u[1] = bp0[1];
        b1.u[0] = bp1[0]; b1.u[1] = bp1[1];

        acc0 = __builtin_amdgcn_wmma_f32_16x16x32_bf16(false, a.v, false, b0.v,
                                                       (short)0, acc0, false, false);
        acc1 = __builtin_amdgcn_wmma_f32_16x16x32_bf16(false, a.v, false, b1.v,
                                                       (short)0, acc1, false, false);
    }

    // epilogue: D[m][n] with m = sel*8 + r, n = tile*16 + l15
    float*          out_f = (float*)hout_v;
    unsigned short* out_b = (unsigned short*)hout_v;
#pragma unroll
    for (int ti = 0; ti < 2; ++ti) {
        v8f acc = ti ? acc1 : acc0;
        int n = (ti ? t1 : t0) * 16 + l15;
        if (n < DOUT) {
            float bv = bias[n];
#pragma unroll
            for (int r = 0; r < 8; ++r) {
                int m = sel * 8 + r;
                float v = acc[r] + bv;
                if (RELU) v = fmaxf(v, 0.0f);
                size_t off = (size_t)(base + m) * DOUT + n;
                if (OUT_F32) out_f[off] = v;
                else         out_b[off] = f32_to_bf16(v);
            }
        }
    }
}

extern "C" void kernel_launch(void* const* d_in, const int* in_sizes, int n_in,
                              void* d_out, int out_size, void* d_ws, size_t ws_size,
                              hipStream_t stream)
{
    const float* x       = (const float*)d_in[0];
    const int*   neigh0  = (const int*)d_in[1];
    const int*   neigh1  = (const int*)d_in[2];
    const int*   neigh2  = (const int*)d_in[3];
    const float* Wself0  = (const float*)d_in[4];
    const float* Wneigh0 = (const float*)d_in[5];
    const float* b0      = (const float*)d_in[6];
    const float* Wself1  = (const float*)d_in[7];
    const float* Wneigh1 = (const float*)d_in[8];
    const float* b1      = (const float*)d_in[9];
    const float* Wself2  = (const float*)d_in[10];
    const float* Wneigh2 = (const float*)d_in[11];
    const float* b2      = (const float*)d_in[12];

    // workspace layout (all 16B aligned)
    char* ws = (char*)d_ws;
    size_t off = 0;
    unsigned short* h1  = (unsigned short*)(ws + off); off += (size_t)67584 * 256 * 2; // bf16 L0 out
    unsigned short* h2  = (unsigned short*)(ws + off); off += (size_t)6144  * 256 * 2; // bf16 L1 out
    unsigned short* wf0 = (unsigned short*)(ws + off); off += (size_t)8  * NT * 512 * 2; // K=256
    unsigned short* wf1 = (unsigned short*)(ws + off); off += (size_t)16 * NT * 512 * 2; // K=512
    unsigned short* wf2 = (unsigned short*)(ws + off); off += (size_t)16 * NT * 512 * 2; // K=512 (N padded)

    // pack weights into WMMA B-fragment layout (bf16)
    pack_weights_kernel<<<64,  256, 0, stream>>>(Wself0, Wneigh0, wf0, 128, 256);
    pack_weights_kernel<<<128, 256, 0, stream>>>(Wself1, Wneigh1, wf1, 256, 256);
    pack_weights_kernel<<<128, 256, 0, stream>>>(Wself2, Wneigh2, wf2, 256, 47);

    // layer 0: fp32 x -> bf16 h1, relu
    sage_layer_kernel<128, 256, 15, false, false, true><<<67584 / 16, TPB, 0, stream>>>(
        x,  neigh0, wf0, b0, h1);
    // layer 1: bf16 h1 -> bf16 h2, relu
    sage_layer_kernel<256, 256, 10, true, false, true><<<6144 / 16, TPB, 0, stream>>>(
        h1, neigh1, wf1, b1, h2);
    // layer 2: bf16 h2 -> fp32 d_out (1024 x 47), no relu
    sage_layer_kernel<256, 47, 5, true, true, false><<<1024 / 16, TPB, 0, stream>>>(
        h2, neigh2, wf2, b2, d_out);
}
